// CrossAttention_86045374808536
// MI455X (gfx1250) — matrix operand order repaired and checked by reading, not compiled, not run on hardware
//
#include <hip/hip_runtime.h>
#include <hip/hip_bf16.h>

#define DMODEL 1024
#define NSEQ   4096

typedef __attribute__((ext_vector_type(16))) _Float16 v16h;
typedef __attribute__((ext_vector_type(8)))  _Float16 v8h;
typedef __attribute__((ext_vector_type(8)))  float    v8f;

union H16 { v16h v; v8h h[2]; };

// Load 8 contiguous f32 and convert to 8 f16 (two b128 loads + pack).
__device__ __forceinline__ v8h load_cvt8(const float* __restrict__ p) {
  const float4 x = *reinterpret_cast<const float4*>(p);
  const float4 y = *reinterpret_cast<const float4*>(p + 4);
  v8h r;
  r[0] = (_Float16)x.x; r[1] = (_Float16)x.y; r[2] = (_Float16)x.z; r[3] = (_Float16)x.w;
  r[4] = (_Float16)y.x; r[5] = (_Float16)y.y; r[6] = (_Float16)y.z; r[7] = (_Float16)y.w;
  return r;
}

__device__ __forceinline__ v16h load16h(const _Float16* __restrict__ p) {
  H16 u;
  u.h[0] = *reinterpret_cast<const v8h*>(p);
  u.h[1] = *reinterpret_cast<const v8h*>(p + 16);
  return u.v;
}

// ---------------------------------------------------------------------------
// Elementwise f32 -> f16 (for weights): one-time, removes cvt chains from the
// projection GEMM's inner loop.
// ---------------------------------------------------------------------------
__global__ __launch_bounds__(256) void cvt_f32_f16(
    const float* __restrict__ src, _Float16* __restrict__ dst, int n8)
{
  const int i = blockIdx.x * 256 + threadIdx.x;
  if (i < n8)
    *reinterpret_cast<v8h*>(dst + (size_t)i * 8) = load_cvt8(src + (size_t)i * 8);
}

// ---------------------------------------------------------------------------
// Projection GEMM: Y = X @ W^T + b   (X:[N,D] f32, Wh:[D,D] f16, b:[D] f32)
// Output f16, optionally transposed ([D,N]) for the V projections so the
// attention P@V B-fragments become contiguous loads.
// Block = 256 threads = 8 waves; wave tile = 16(M) x 64(N); block = 32 x 256.
// Ping-pong pipelined k-loop (peeled tail): no per-iteration branches and no
// register-rotation copies.
// ---------------------------------------------------------------------------
__global__ __launch_bounds__(256) void proj_gemm_f16(
    const float* __restrict__ X, const _Float16* __restrict__ Wh,
    const float* __restrict__ bias, _Float16* __restrict__ Y,
    int transpose_out)
{
  const int lane = threadIdx.x & 31;
  const int wave = threadIdx.x >> 5;
  const int ln   = lane & 15;
  const int kg   = lane >> 4;

  const int Mbase = blockIdx.y * 32  + (wave & 1) * 16;
  const int Nbase = blockIdx.x * 256 + (wave >> 1) * 64;

  const float*    arow = X + (size_t)(Mbase + ln) * DMODEL + 8 * kg;
  const _Float16* wrow[4];
#pragma unroll
  for (int t = 0; t < 4; ++t)
    wrow[t] = Wh + (size_t)(Nbase + t * 16 + ln) * DMODEL + 8 * kg;

  v8f acc[4] = {};
  H16  a0, a1;
  v16h b0[4], b1[4];

  // prologue: fragments for k0 = 0
  a0.h[0] = load_cvt8(arow);
  a0.h[1] = load_cvt8(arow + 16);
#pragma unroll
  for (int t = 0; t < 4; ++t) b0[t] = load16h(wrow[t]);

  for (int k0 = 0; k0 < DMODEL - 64; k0 += 64) {
    // prefetch k0+32 into buffer 1
    a1.h[0] = load_cvt8(arow + k0 + 32);
    a1.h[1] = load_cvt8(arow + k0 + 48);
#pragma unroll
    for (int t = 0; t < 4; ++t) b1[t] = load16h(wrow[t] + k0 + 32);
#pragma unroll
    for (int t = 0; t < 4; ++t)
      acc[t] = __builtin_amdgcn_wmma_f32_16x16x32_f16(
          false, a0.v, false, b0[t], (short)0, acc[t], false, false);
    // prefetch k0+64 into buffer 0
    a0.h[0] = load_cvt8(arow + k0 + 64);
    a0.h[1] = load_cvt8(arow + k0 + 80);
#pragma unroll
    for (int t = 0; t < 4; ++t) b0[t] = load16h(wrow[t] + k0 + 64);
#pragma unroll
    for (int t = 0; t < 4; ++t)
      acc[t] = __builtin_amdgcn_wmma_f32_16x16x32_f16(
          false, a1.v, false, b1[t], (short)0, acc[t], false, false);
  }
  // peeled tail: buffer 0 holds k0 = DMODEL-64; load DMODEL-32 into buffer 1
  a1.h[0] = load_cvt8(arow + DMODEL - 32);
  a1.h[1] = load_cvt8(arow + DMODEL - 16);
#pragma unroll
  for (int t = 0; t < 4; ++t) b1[t] = load16h(wrow[t] + DMODEL - 32);
#pragma unroll
  for (int t = 0; t < 4; ++t)
    acc[t] = __builtin_amdgcn_wmma_f32_16x16x32_f16(
        false, a0.v, false, b0[t], (short)0, acc[t], false, false);
#pragma unroll
  for (int t = 0; t < 4; ++t)
    acc[t] = __builtin_amdgcn_wmma_f32_16x16x32_f16(
        false, a1.v, false, b1[t], (short)0, acc[t], false, false);

#pragma unroll
  for (int t = 0; t < 4; ++t) {
    const int col = Nbase + t * 16 + ln;
    const float bv = bias[col];
#pragma unroll
    for (int r = 0; r < 8; ++r) {
      const int   m   = Mbase + r + 8 * kg;      // C layout: M = r + 8*(lane>>4)
      const float val = acc[t][r] + bv;
      if (transpose_out) Y[(size_t)col * NSEQ + m]    = (_Float16)val;
      else               Y[(size_t)m  * DMODEL + col] = (_Float16)val;
    }
  }
}

// ---------------------------------------------------------------------------
// Flash cross-attention: O = softmax(Q @ K^T / sqrt(D)) @ V
// Q:[N,D] f16, K:[N,D] f16, VT:[D,N] f16, O:[N,D] f32.
// Block = 256 threads = 8 waves, 16 query rows per block.
// ---------------------------------------------------------------------------
#define QPAD 1032   // 16B-aligned, 4-bank row skew for conflict-free ds_load_b128
#define SPAD 136    // 16B-aligned f16 rows, skewed banks

__global__ __launch_bounds__(256) void flash_attn_f16(
    const _Float16* __restrict__ Q, const _Float16* __restrict__ K,
    const _Float16* __restrict__ VT, float* __restrict__ O)
{
  __shared__ _Float16 q_s[16 * QPAD];     // Q tile, f16
  __shared__ float    s_s[16 * SPAD];     // raw scores
  __shared__ _Float16 p_s[16 * SPAD];     // probabilities, f16
  __shared__ float    m_s[16], l_s[16], al_s[16];

  const int tid   = threadIdx.x;
  const int lane  = tid & 31;
  const int wave  = tid >> 5;
  const int ln    = lane & 15;
  const int kg    = lane >> 4;
  const int qbase = blockIdx.x * 16;
  const int dbase = wave * 128;

  // Stage Q tile into LDS (global_load_b128 + ds_store_b128)
  for (int j = tid; j < 16 * (DMODEL / 8); j += 256) {
    const int row = j >> 7;            // DMODEL/8 = 128 chunks per row
    const int c   = (j & 127) * 8;
    *reinterpret_cast<v8h*>(q_s + row * QPAD + c) =
        *reinterpret_cast<const v8h*>(Q + (size_t)(qbase + row) * DMODEL + c);
  }
  if (tid < 16) { m_s[tid] = -1e30f; l_s[tid] = 0.f; }
  __syncthreads();

  v8f acc[8] = {};
  const float scale = 1.0f / 32.0f;    // 1/sqrt(1024)

  for (int jb = 0; jb < NSEQ / 128; ++jb) {
    const int keybase = jb * 128;

    // ---- score phase: S[16 x 16] per wave, ping-pong pipelined B loads ----
    const int keycol = keybase + wave * 16 + ln;
    const _Float16* krow = K + (size_t)keycol * DMODEL + 8 * kg;
    const _Float16* qrow = q_s + ln * QPAD + 8 * kg;
    if (jb + 1 < NSEQ / 128)           // pull next key block toward L2/L0
      __builtin_prefetch(K + (size_t)(keycol + 128) * DMODEL, 0, 1);

    v8f  sacc = {};
    v16h b0 = load16h(krow), b1;
    H16  a;
    for (int k0 = 0; k0 < DMODEL - 64; k0 += 64) {
      b1 = load16h(krow + k0 + 32);
      a.h[0] = *reinterpret_cast<const v8h*>(qrow + k0);
      a.h[1] = *reinterpret_cast<const v8h*>(qrow + k0 + 16);
      sacc = __builtin_amdgcn_wmma_f32_16x16x32_f16(
          false, a.v, false, b0, (short)0, sacc, false, false);
      b0 = load16h(krow + k0 + 64);
      a.h[0] = *reinterpret_cast<const v8h*>(qrow + k0 + 32);
      a.h[1] = *reinterpret_cast<const v8h*>(qrow + k0 + 48);
      sacc = __builtin_amdgcn_wmma_f32_16x16x32_f16(
          false, a.v, false, b1, (short)0, sacc, false, false);
    }
    b1 = load16h(krow + DMODEL - 32);
    a.h[0] = *reinterpret_cast<const v8h*>(qrow + DMODEL - 64);
    a.h[1] = *reinterpret_cast<const v8h*>(qrow + DMODEL - 48);
    sacc = __builtin_amdgcn_wmma_f32_16x16x32_f16(
        false, a.v, false, b0, (short)0, sacc, false, false);
    a.h[0] = *reinterpret_cast<const v8h*>(qrow + DMODEL - 32);
    a.h[1] = *reinterpret_cast<const v8h*>(qrow + DMODEL - 16);
    sacc = __builtin_amdgcn_wmma_f32_16x16x32_f16(
        false, a.v, false, b1, (short)0, sacc, false, false);

#pragma unroll
    for (int r = 0; r < 8; ++r)
      s_s[(r + 8 * kg) * SPAD + wave * 16 + ln] = sacc[r] * scale;
    __syncthreads();

    // ---- online softmax: 16 threads per row, shfl_xor reductions ----
    {
      const int row = tid >> 4, sub = tid & 15;
      const float mold = m_s[row];
      float v[8], bmax = -1e30f;
#pragma unroll
      for (int i = 0; i < 8; ++i) {
        v[i] = s_s[row * SPAD + sub * 8 + i];
        bmax = fmaxf(bmax, v[i]);
      }
#pragma unroll
      for (int off = 8; off >= 1; off >>= 1)
        bmax = fmaxf(bmax, __shfl_xor(bmax, off, 16));
      const float mnew = fmaxf(mold, bmax);
      float sum = 0.f;
#pragma unroll
      for (int i = 0; i < 8; ++i) {
        const float p = __expf(v[i] - mnew);
        sum += p;
        p_s[row * SPAD + sub * 8 + i] = (_Float16)p;
      }
#pragma unroll
      for (int off = 8; off >= 1; off >>= 1)
        sum += __shfl_xor(sum, off, 16);
      if (sub == 0) {
        const float alpha = __expf(mold - mnew);
        l_s[row]  = l_s[row] * alpha + sum;
        m_s[row]  = mnew;
        al_s[row] = alpha;
      }
    }
    __syncthreads();

    // ---- rescale running accumulators by alpha[row] ----
    float alr[8];
#pragma unroll
    for (int r = 0; r < 8; ++r) alr[r] = al_s[r + 8 * kg];
#pragma unroll
    for (int t = 0; t < 8; ++t)
#pragma unroll
      for (int r = 0; r < 8; ++r) acc[t][r] *= alr[r];

    // ---- P @ V: batch the 8 B-fragment loads, then 8 WMMAs ----
    for (int kk0 = 0; kk0 < 128; kk0 += 32) {
      H16 pa;
      pa.h[0] = *reinterpret_cast<const v8h*>(p_s + ln * SPAD + kk0 + 8 * kg);
      pa.h[1] = *reinterpret_cast<const v8h*>(p_s + ln * SPAD + kk0 + 16 + 8 * kg);
      v16h bb[8];
#pragma unroll
      for (int t = 0; t < 8; ++t)
        bb[t] = load16h(VT + (size_t)(dbase + t * 16 + ln) * NSEQ +
                        keybase + kk0 + 8 * kg);
#pragma unroll
      for (int t = 0; t < 8; ++t)
        acc[t] = __builtin_amdgcn_wmma_f32_16x16x32_f16(
            false, pa.v, false, bb[t], (short)0, acc[t], false, false);
    }
    __syncthreads();
  }

  // ---- epilogue: O = acc / l ----
  float linv[8];
#pragma unroll
  for (int r = 0; r < 8; ++r) linv[r] = 1.0f / l_s[r + 8 * kg];
#pragma unroll
  for (int t = 0; t < 8; ++t) {
    const int dcol = dbase + t * 16 + ln;
#pragma unroll
    for (int r = 0; r < 8; ++r) {
      const int m = qbase + r + 8 * kg;
      O[(size_t)m * DMODEL + dcol] = acc[t][r] * linv[r];
    }
  }
}

// ---------------------------------------------------------------------------
extern "C" void kernel_launch(void* const* d_in, const int* in_sizes, int n_in,
                              void* d_out, int out_size, void* d_ws, size_t ws_size,
                              hipStream_t stream)
{
  (void)in_sizes; (void)n_in; (void)out_size; (void)ws_size;

  // setup_inputs() dict order: 6 tensors, then interleaved W/b pairs.
  const float* Qc   = (const float*)d_in[0];
  const float* Kc   = (const float*)d_in[1];
  const float* Vc   = (const float*)d_in[2];
  const float* Qp   = (const float*)d_in[3];
  const float* Kp   = (const float*)d_in[4];
  const float* Vp   = (const float*)d_in[5];
  const float* Wq_c = (const float*)d_in[6];  const float* bq_c = (const float*)d_in[7];
  const float* Wk_c = (const float*)d_in[8];  const float* bk_c = (const float*)d_in[9];
  const float* Wv_c = (const float*)d_in[10]; const float* bv_c = (const float*)d_in[11];
  const float* Wq_p = (const float*)d_in[12]; const float* bq_p = (const float*)d_in[13];
  const float* Wk_p = (const float*)d_in[14]; const float* bk_p = (const float*)d_in[15];
  const float* Wv_p = (const float*)d_in[16]; const float* bv_p = (const float*)d_in[17];

  const size_t NE = (size_t)NSEQ * DMODEL;       // 4M elements
  const size_t DD = (size_t)DMODEL * DMODEL;     // 1M elements

  // Workspace layout: 6 f16 weights (12 MB) + 6 f16 q/k/vT (48 MB) = 60 MB.
  _Float16* whq_c = (_Float16*)d_ws;
  _Float16* whk_c = whq_c + DD;
  _Float16* whv_c = whk_c + DD;
  _Float16* whq_p = whv_c + DD;
  _Float16* whk_p = whq_p + DD;
  _Float16* whv_p = whk_p + DD;
  _Float16* qc  = whv_p + DD;
  _Float16* kc  = qc  + NE;
  _Float16* vTc = kc  + NE;          // transposed [D, N]
  _Float16* qp  = vTc + NE;
  _Float16* kp  = qp  + NE;
  _Float16* vTp = kp  + NE;          // transposed [D, N]

  const dim3 pb(256);
  const int  cvtg = (int)(DD / 8 / 256);         // 512 blocks
  cvt_f32_f16<<<dim3(cvtg), pb, 0, stream>>>(Wq_c, whq_c, (int)(DD / 8));
  cvt_f32_f16<<<dim3(cvtg), pb, 0, stream>>>(Wk_c, whk_c, (int)(DD / 8));
  cvt_f32_f16<<<dim3(cvtg), pb, 0, stream>>>(Wv_c, whv_c, (int)(DD / 8));
  cvt_f32_f16<<<dim3(cvtg), pb, 0, stream>>>(Wq_p, whq_p, (int)(DD / 8));
  cvt_f32_f16<<<dim3(cvtg), pb, 0, stream>>>(Wk_p, whk_p, (int)(DD / 8));
  cvt_f32_f16<<<dim3(cvtg), pb, 0, stream>>>(Wv_p, whv_p, (int)(DD / 8));

  const dim3 pg(DMODEL / 256, NSEQ / 32);
  proj_gemm_f16<<<pg, pb, 0, stream>>>(Qc, whq_c, bq_c, qc,  0);
  proj_gemm_f16<<<pg, pb, 0, stream>>>(Kc, whk_c, bk_c, kc,  0);
  proj_gemm_f16<<<pg, pb, 0, stream>>>(Vc, whv_c, bv_c, vTc, 1);
  proj_gemm_f16<<<pg, pb, 0, stream>>>(Qp, whq_p, bq_p, qp,  0);
  proj_gemm_f16<<<pg, pb, 0, stream>>>(Kp, whk_p, bk_p, kp,  0);
  proj_gemm_f16<<<pg, pb, 0, stream>>>(Vp, whv_p, bv_p, vTp, 1);

  float* comp = (float*)d_out;       // [Nc, D]
  float* prot = comp + NE;           // [Np, D]
  flash_attn_f16<<<dim3(NSEQ / 16), pb, 0, stream>>>(qc, kp, vTp, comp);
  flash_attn_f16<<<dim3(NSEQ / 16), pb, 0, stream>>>(qp, kc, vTc, prot);
}
